// Net_25426206392783
// MI455X (gfx1250) — matrix operand order repaired and checked by reading, not compiled, not run on hardware
//
#include <hip/hip_runtime.h>
#include <hip/hip_bf16.h>
#include <cstddef>

// ---------------- problem constants (from reference) ----------------
static constexpr int NB    = 8;      // graphs
static constexpr int NPTS  = 2048;   // points per graph
static constexpr int TOT   = NB * NPTS;
static constexpr int KNN   = 20;     // neighbors
static constexpr int CHUNKS = NPTS / 32; // 64 point-chunks for global max

typedef _Float16 half_t;
typedef __attribute__((ext_vector_type(16))) _Float16 v16h;
typedef __attribute__((ext_vector_type(8)))  float    v8f;

// ---------------- WMMA helpers (v_wmma_f32_16x16x32_f16) ----------------
// A-matrix 16x32 f16 layout (ISA 7.12.2): lane = 16*hi + m (m = M row 0..15).
// VGPR pair p (halves 2p,2p+1): K = (p<4 ? 2p : 16+2(p-4)) + h + (hi ? 8 : 0).
__device__ __forceinline__ v16h load_a_tile(const _Float16* feat, int kpad,
                                            int mbase, int kbase) {
  int lane = threadIdx.x & 31;
  int m  = mbase + (lane & 15);
  int hi = lane >> 4;
  v16h a;
#pragma unroll
  for (int p = 0; p < 8; ++p) {
    int k = kbase + ((p < 4) ? (2 * p) : (16 + 2 * (p - 4))) + (hi ? 8 : 0);
    a[2 * p]     = feat[m * kpad + k];
    a[2 * p + 1] = feat[m * kpad + k + 1];
  }
  return a;
}

// B-matrix 32x16 f16 layout: lane = 16*hi + n; half e in lane holds
// K = (e<8 ? e : 8+e) + 8*hi, N = nbase + (lane&15).  We PRE-PACK weights into this exact
// per-lane layout so the hot loop is a single aligned 32B vector load.
// packed[((nt*nkb + kb)*32 + lane)*16 + e]
__global__ void pack_b_kernel(const float* __restrict__ W,
                              _Float16* __restrict__ packed,
                              int ktot, int cout, int nkb) {
  const int total = (cout / 16) * nkb * 32 * 16;
  const int gid = blockIdx.x * blockDim.x + threadIdx.x;
  if (gid >= total) return;
  const int e    = gid & 15;
  const int lane = (gid >> 4) & 31;
  const int tile = gid >> 9;            // nt*nkb + kb
  const int kb   = tile % nkb;
  const int nt   = tile / nkb;
  const int n = nt * 16 + (lane & 15);
  const int k = kb * 32 + ((e < 8) ? e : (8 + e)) + ((lane >> 4) ? 8 : 0);
  packed[gid] = (k < ktot) ? (_Float16)W[(size_t)k * cout + n] : (_Float16)0.f;
}

__device__ __forceinline__ v16h load_b_packed(const _Float16* __restrict__ packed,
                                              int nt, int nkb, int kb32) {
  const int lane = threadIdx.x & 31;
  const size_t off = ((size_t)((nt * nkb + kb32) * 32 + lane)) << 4;
  return *(const v16h*)(packed + off);
}

__device__ __forceinline__ v8f wmma_f16(v16h a, v16h b, v8f c) {
  // D = A(16x32) * B(32x16) + C, f32 accumulate
  return __builtin_amdgcn_wmma_f32_16x16x32_f16(false, a, false, b,
                                                (short)0, c, false, false);
}

// ---------------- kNN: brute force, LDS-tiled, register top-K ----------------
template <int C, int TILE>
__global__ void knn_kernel(const float* __restrict__ x, int* __restrict__ idx) {
  __shared__ float xs[TILE * C];
  const int b = blockIdx.y;
  const float* xb = x + (size_t)b * NPTS * C;
  const int i = blockIdx.x * blockDim.x + threadIdx.x;

  float xi[C];
#pragma unroll
  for (int c = 0; c < C; ++c) xi[c] = xb[(size_t)i * C + c];

  float bestd[KNN];
  int   bestj[KNN];
#pragma unroll
  for (int t = 0; t < KNN; ++t) { bestd[t] = 3.0e38f; bestj[t] = 0; }
  float worstd = 3.0e38f;
  int   worstpos = 0;

  for (int t0 = 0; t0 < NPTS; t0 += TILE) {
    __syncthreads();
    for (int e = threadIdx.x; e < TILE * C; e += blockDim.x)
      xs[e] = xb[(size_t)t0 * C + e];
    __syncthreads();
    for (int jj = 0; jj < TILE; ++jj) {
      float d2 = 0.f;
#pragma unroll
      for (int c = 0; c < C; ++c) {
        float d = xi[c] - xs[jj * C + c];
        d2 = fmaf(d, d, d2);
      }
      if (d2 < worstd) {
        const int j = t0 + jj;
#pragma unroll
        for (int t = 0; t < KNN; ++t)
          if (t == worstpos) { bestd[t] = d2; bestj[t] = j; }
        worstd = -3.0e38f;
#pragma unroll
        for (int t = 0; t < KNN; ++t)
          if (bestd[t] > worstd) { worstd = bestd[t]; worstpos = t; }
      }
    }
  }
#pragma unroll
  for (int t = 0; t < KNN; ++t)
    idx[(size_t)(b * NPTS + i) * KNN + t] = b * NPTS + bestj[t];
}

// ---------------- EdgeConv: WMMA message GEMM fused with max over K --------
// One block per point. feat = [x_i, x_j - x_i] for the 20 edges, padded to 32
// rows (pad rows duplicate edge 0 -> max unaffected) and K padded to mult of 32.
// Each wave computes one 16-channel output tile with two 16x16x32 WMMA accs.
template <int CIN, int COUT>
__global__ void edgeconv_kernel(const float* __restrict__ x,
                                const int* __restrict__ idx,
                                const _Float16* __restrict__ pb, // packed W
                                const float* __restrict__ bias,
                                float* __restrict__ out) {
  constexpr int KTOT = 2 * CIN;
  constexpr int KPAD = (KTOT + 31) & ~31;
  constexpr int NKB  = KPAD / 32;
  __shared__ _Float16 feat[32 * KPAD];
  __shared__ int   jrow[32];
  __shared__ float xisl[CIN];

  const int pt  = blockIdx.x;
  const int tid = threadIdx.x;

  // stage neighbor row indices + this point's own features once
  if (tid < 32) {
    const int e = (tid < KNN) ? tid : 0;
    jrow[tid] = idx[(size_t)pt * KNN + e];
  }
  for (int c = tid; c < CIN; c += blockDim.x)
    xisl[c] = x[(size_t)pt * CIN + c];
  __syncthreads();

  for (int t = tid; t < 32 * KPAD; t += blockDim.x) {
    const int r = t / KPAD, c = t % KPAD;
    float v;
    if (c < CIN) {
      v = xisl[c];
    } else if (c < KTOT) {
      const int j = jrow[r];
      v = x[(size_t)j * CIN + (c - CIN)] - xisl[c - CIN];
    } else {
      v = 0.f;
    }
    feat[r * KPAD + c] = (_Float16)v;
  }
  __syncthreads();

  const int wave = tid >> 5;   // = output n-tile
  const int lane = tid & 31;
  v8f acc0 = {};
  v8f acc1 = {};
#pragma unroll
  for (int kb = 0; kb < NKB; ++kb) {
    if (kb + 1 < NKB)  // prefetch next packed-B tile (global_prefetch_b8)
      __builtin_prefetch(pb + (((size_t)(wave * NKB + kb + 1) * 32 + lane) << 4), 0, 3);
    v16h a0 = load_a_tile(feat, KPAD, 0,  kb * 32);
    v16h a1 = load_a_tile(feat, KPAD, 16, kb * 32);
    v16h bb = load_b_packed(pb, wave, NKB, kb);
    acc0 = wmma_f16(a0, bb, acc0);
    acc1 = wmma_f16(a1, bb, acc1);
  }
  // max over M rows: VGPR r holds M=r (lanes 0-15) / M=8+r (lanes 16-31)
  float m = -3.0e38f;
#pragma unroll
  for (int r = 0; r < 8; ++r) m = fmaxf(m, fmaxf(acc0[r], acc1[r]));
  m = fmaxf(m, __shfl_xor(m, 16, 32));
  if (lane < 16) {
    const int ch = wave * 16 + lane;
    out[(size_t)pt * COUT + ch] = m + bias[ch];
  }
}

// ---------------- concat @ lw fused with per-chunk max over points ----------
// Block: (graph, 32-point chunk, 128-channel group). Writes partial maxima.
__global__ void lastlin_kernel(const float* __restrict__ x1,
                               const float* __restrict__ x2,
                               const float* __restrict__ x3,
                               const float* __restrict__ x4,
                               const _Float16* __restrict__ plw, // packed lw
                               float* __restrict__ pmax) {
  constexpr int KTOT = 512;
  constexpr int NKB  = KTOT / 32;
  __shared__ _Float16 feat[32 * KTOT];
  const int g  = blockIdx.z;
  const int ch = blockIdx.y;
  const int ng = blockIdx.x;
  const int tid = threadIdx.x;
  const size_t prow0 = (size_t)g * NPTS + ch * 32;

  for (int t = tid; t < 32 * KTOT; t += blockDim.x) {
    const int r = t / KTOT, c = t % KTOT;
    const size_t p = prow0 + r;
    float v;
    if (c < 64)        v = x1[p * 64 + c];
    else if (c < 128)  v = x2[p * 64 + (c - 64)];
    else if (c < 256)  v = x3[p * 128 + (c - 128)];
    else               v = x4[p * 256 + (c - 256)];
    feat[r * KTOT + c] = (_Float16)v;
  }
  __syncthreads();

  const int wave = tid >> 5;
  const int lane = tid & 31;
  const int nt   = ng * 8 + wave;   // n-tile 0..63
  v8f acc0 = {};
  v8f acc1 = {};
  for (int kb = 0; kb < NKB; ++kb) {
    if (kb + 1 < NKB)
      __builtin_prefetch(plw + (((size_t)(nt * NKB + kb + 1) * 32 + lane) << 4), 0, 3);
    v16h a0 = load_a_tile(feat, KTOT, 0,  kb * 32);
    v16h a1 = load_a_tile(feat, KTOT, 16, kb * 32);
    v16h bb = load_b_packed(plw, nt, NKB, kb);
    acc0 = wmma_f16(a0, bb, acc0);
    acc1 = wmma_f16(a1, bb, acc1);
  }
  float m = -3.0e38f;
#pragma unroll
  for (int r = 0; r < 8; ++r) m = fmaxf(m, fmaxf(acc0[r], acc1[r]));
  m = fmaxf(m, __shfl_xor(m, 16, 32));
  if (lane < 16)
    pmax[((size_t)g * CHUNKS + ch) * 1024 + nt * 16 + lane] = m;
}

__global__ void gmax_kernel(const float* __restrict__ pmax,
                            const float* __restrict__ lb,
                            float* __restrict__ gout) {
  const int g = blockIdx.x, c = threadIdx.x;
  float m = -3.0e38f;
  for (int ch = 0; ch < CHUNKS; ++ch)
    m = fmaxf(m, pmax[((size_t)g * CHUNKS + ch) * 1024 + c]);
  gout[(size_t)g * 1024 + c] = m + lb[c];
}

// ---------------- tiny MLP head + log_softmax (one block per graph) --------
__global__ void mlp_kernel(const float* __restrict__ gin,
                           const float* __restrict__ m1w, const float* __restrict__ m1b,
                           const float* __restrict__ m2w, const float* __restrict__ m2b,
                           const float* __restrict__ m3w, const float* __restrict__ m3b,
                           float* __restrict__ out) {
  __shared__ float gb[1024];
  __shared__ float h1[512];
  __shared__ float h2[256];
  __shared__ float lg[10];
  __shared__ float red[2];
  const int g = blockIdx.x, t = threadIdx.x;

  for (int i = t; i < 1024; i += blockDim.x) gb[i] = gin[(size_t)g * 1024 + i];
  __syncthreads();
  if (t < 512) {
    float s = m1b[t];
    for (int i = 0; i < 1024; ++i) s = fmaf(gb[i], m1w[(size_t)i * 512 + t], s);
    h1[t] = fmaxf(s, 0.f);
  }
  __syncthreads();
  if (t < 256) {
    float s = m2b[t];
    for (int i = 0; i < 512; ++i) s = fmaf(h1[i], m2w[(size_t)i * 256 + t], s);
    h2[t] = fmaxf(s, 0.f);
  }
  __syncthreads();
  if (t < 10) {
    float s = m3b[t];
    for (int i = 0; i < 256; ++i) s = fmaf(h2[i], m3w[(size_t)i * 10 + t], s);
    lg[t] = s;
  }
  __syncthreads();
  if (t == 0) {
    float mx = lg[0];
    for (int i = 1; i < 10; ++i) mx = fmaxf(mx, lg[i]);
    float se = 0.f;
    for (int i = 0; i < 10; ++i) se += __expf(lg[i] - mx);
    red[0] = mx;
    red[1] = __logf(se);
  }
  __syncthreads();
  if (t < 10) out[(size_t)g * 10 + t] = lg[t] - red[0] - red[1];
}

// ---------------- launch ----------------
static inline size_t al256(size_t x) { return (x + 255) & ~(size_t)255; }
static inline int packed_halves(int cout, int nkb) { return (cout / 16) * nkb * 32 * 16; }

extern "C" void kernel_launch(void* const* d_in, const int* in_sizes, int n_in,
                              void* d_out, int out_size, void* d_ws, size_t ws_size,
                              hipStream_t stream) {
  const float* pos = (const float*)d_in[0];
  const float* w1  = (const float*)d_in[1];
  const float* b1  = (const float*)d_in[2];
  const float* w2  = (const float*)d_in[3];
  const float* b2  = (const float*)d_in[4];
  const float* w3  = (const float*)d_in[5];
  const float* b3  = (const float*)d_in[6];
  const float* w4  = (const float*)d_in[7];
  const float* b4  = (const float*)d_in[8];
  const float* lw  = (const float*)d_in[9];
  const float* lb  = (const float*)d_in[10];
  const float* m1w = (const float*)d_in[11];
  const float* m1b = (const float*)d_in[12];
  const float* m2w = (const float*)d_in[13];
  const float* m2b = (const float*)d_in[14];
  const float* m3w = (const float*)d_in[15];
  const float* m3b = (const float*)d_in[16];
  float* out = (float*)d_out;

  // workspace carve-out
  char* ws = (char*)d_ws;
  size_t o = 0;
  int*   idx  = (int*)(ws + o);   o += al256((size_t)TOT * KNN * sizeof(int));
  float* x1   = (float*)(ws + o); o += al256((size_t)TOT * 64  * sizeof(float));
  float* x2   = (float*)(ws + o); o += al256((size_t)TOT * 64  * sizeof(float));
  float* x3   = (float*)(ws + o); o += al256((size_t)TOT * 128 * sizeof(float));
  float* x4   = (float*)(ws + o); o += al256((size_t)TOT * 256 * sizeof(float));
  float* pmax = (float*)(ws + o); o += al256((size_t)NB * CHUNKS * 1024 * sizeof(float));
  float* gvec = (float*)(ws + o); o += al256((size_t)NB * 1024 * sizeof(float));
  _Float16* pw1 = (_Float16*)(ws + o); o += al256((size_t)packed_halves(64,   1) * 2);
  _Float16* pw2 = (_Float16*)(ws + o); o += al256((size_t)packed_halves(64,   4) * 2);
  _Float16* pw3 = (_Float16*)(ws + o); o += al256((size_t)packed_halves(128,  4) * 2);
  _Float16* pw4 = (_Float16*)(ws + o); o += al256((size_t)packed_halves(256,  8) * 2);
  _Float16* plw = (_Float16*)(ws + o); o += al256((size_t)packed_halves(1024, 16) * 2);
  (void)ws_size; (void)in_sizes; (void)n_in; (void)out_size;

  // pre-pack all GEMM weights into WMMA B-register layout (f16), once per call
  {
    int n;
    n = packed_halves(64, 1);
    pack_b_kernel<<<(n + 255) / 256, 256, 0, stream>>>(w1, pw1, 6,   64,   1);
    n = packed_halves(64, 4);
    pack_b_kernel<<<(n + 255) / 256, 256, 0, stream>>>(w2, pw2, 128, 64,   4);
    n = packed_halves(128, 4);
    pack_b_kernel<<<(n + 255) / 256, 256, 0, stream>>>(w3, pw3, 128, 128,  4);
    n = packed_halves(256, 8);
    pack_b_kernel<<<(n + 255) / 256, 256, 0, stream>>>(w4, pw4, 256, 256,  8);
    n = packed_halves(1024, 16);
    pack_b_kernel<<<(n + 255) / 256, 256, 0, stream>>>(lw, plw, 512, 1024, 16);
  }

  const dim3 kg(NPTS / 128, NB);
  const dim3 kb(128);

  // layer 1: kNN on pos(3) -> edgeconv 6->64
  knn_kernel<3, 64><<<kg, kb, 0, stream>>>(pos, idx);
  edgeconv_kernel<3, 64><<<TOT, 32 * (64 / 16), 0, stream>>>(pos, idx, pw1, b1, x1);

  // layer 2: kNN on x1(64) -> edgeconv 128->64
  knn_kernel<64, 64><<<kg, kb, 0, stream>>>(x1, idx);
  edgeconv_kernel<64, 64><<<TOT, 32 * (64 / 16), 0, stream>>>(x1, idx, pw2, b2, x2);

  // layer 3: kNN on x2(64) -> edgeconv 128->128
  knn_kernel<64, 64><<<kg, kb, 0, stream>>>(x2, idx);
  edgeconv_kernel<64, 128><<<TOT, 32 * (128 / 16), 0, stream>>>(x2, idx, pw3, b3, x3);

  // layer 4: kNN on x3(128) -> edgeconv 256->256
  knn_kernel<128, 64><<<kg, kb, 0, stream>>>(x3, idx);
  edgeconv_kernel<128, 256><<<TOT, 32 * (256 / 16), 0, stream>>>(x3, idx, pw4, b4, x4);

  // concat @ lw fused with chunk-max, then global max + lb
  lastlin_kernel<<<dim3(8, CHUNKS, NB), 256, 0, stream>>>(x1, x2, x3, x4, plw, pmax);
  gmax_kernel<<<NB, 1024, 0, stream>>>(pmax, lb, gvec);

  // MLP head + log_softmax
  mlp_kernel<<<NB, 512, 0, stream>>>(gvec, m1w, m1b, m2w, m2b, m3w, m3b, out);
}
